// SE_11218454577173
// MI455X (gfx1250) — compile-verified, matrix operand order checked by reading, not run on hardware
//
#include <hip/hip_runtime.h>
#include <hip/hip_bf16.h>

// ---------------------------------------------------------------------------
// Vision-SSM block for MI455X (gfx1250, wave32).
// GEMMs via v_wmma_f32_16x16x32_f16 (f16 in / f32 accumulate).
// Memory-bound (~20 GFLOP vs ~1.1 GB HBM @ 23.3 TB/s) -> streaming layouts,
// WMMA keeps math off the critical path.  This revision pre-swizzles all
// weights into WMMA fragment order so B-fragment loads are single 32B vector
// loads (no strided scalar gathers, no f32->f16 packing VALU in the hot loop).
// ---------------------------------------------------------------------------

#define DEVINL __device__ __forceinline__

typedef __attribute__((ext_vector_type(16))) _Float16 v16h;
typedef __attribute__((ext_vector_type(8)))  float    v8f;

// Problem constants
#define BSZ   8
#define C1N   64
#define DI    128      // d_inner
#define DST   128      // d_state
#define HH    128
#define WW    128
#define HWP   16384    // H*W (= L per image)
#define NTOK  131072   // B*H*W
#define CHUNK 256      // scan chunk length
#define NCHK  64       // HWP / CHUNK

DEVINL v8f wmma_f16(v16h a, v16h b, v8f c) {
  // D = A(16x32 f16) * B(32x16 f16) + C(16x16 f32)
  return __builtin_amdgcn_wmma_f32_16x16x32_f16(
      /*neg_a=*/false, a, /*neg_b=*/false, b,
      /*c_mod=*/(short)0, c, /*reuse_a=*/false, /*reuse_b=*/false);
}

// A fragment (16x32, f16) from an LDS tile laid out [row][k], leading dim ld.
// ISA layout: lane<16 holds row M=lane, K in {0..7,16..23}; lane>=16 same row,
// K in {8..15,24..31}.  half h -> K = 16*(h>>3) + 8*hi + (h&7).  The two
// 8-half runs are contiguous & 16B aligned -> 2x ds_load_b128 per fragment.
DEVINL v16h frag_a_lds(const _Float16* t, int ld) {
  const int lane = threadIdx.x & 31;
  const int row  = lane & 15;
  const int koff = (lane >> 4) << 3;
  v16h f;
#pragma unroll
  for (int h = 0; h < 16; ++h) {
    const int k = ((h >> 3) << 4) + koff + (h & 7);
    f[h] = t[row * ld + k];
  }
  return f;
}

// B fragment from a pre-swizzled (fragment-order) f16 weight array:
// element (((nt*ksteps + kk)*32 + lane)*16 + h).  One aligned 32B load/lane.
DEVINL v16h frag_b_pre(const _Float16* __restrict__ base, int ksteps,
                       int kk, int nt) {
  const int lane = threadIdx.x & 31;
  return *(const v16h*)(base + (((size_t)(nt * ksteps + kk) * 32 + lane) << 4));
}

DEVINL float silu_f(float v) { return v / (1.f + __expf(-v)); }

// ---------------------------------------------------------------------------
// Kernel 0: swizzle an f32 [K][N] weight matrix into WMMA B-fragment order
// (f16).  i = ((nt*ksteps + kk)*32 + lane)*16 + h.
// ---------------------------------------------------------------------------
__global__ __launch_bounds__(256) void k_wswz(const float* __restrict__ W,
                                              int K, int N,
                                              _Float16* __restrict__ dst) {
  const int ksteps = K >> 5;
  const int i = blockIdx.x * 256 + threadIdx.x;
  if (i >= K * N) return;
  const int h    = i & 15;
  const int lane = (i >> 4) & 31;
  const int rest = i >> 9;                 // nt*ksteps + kk
  const int kk   = rest % ksteps;
  const int nt   = rest / ksteps;
  const int k = kk * 32 + ((h >> 3) << 4) + ((lane >> 4) << 3) + (h & 7);
  const int n = nt * 16 + (lane & 15);
  dst[i] = (_Float16)W[k * N + n];
}

// ---------------------------------------------------------------------------
// Kernel 1: in-projection.  xz = xh @ w_in + b_in ; split into x1 (->NCHW for
// conv) and z (token-major).  M-tile = 128 tokens/block, 8 waves x 16 rows.
// ---------------------------------------------------------------------------
__global__ __launch_bounds__(256) void k_inproj(
    const float* __restrict__ x, const _Float16* __restrict__ wfin,
    const float* __restrict__ b_in, float* __restrict__ x1n,
    float* __restrict__ zbuf) {
  __shared__ _Float16 At[128][72];     // 128 tokens x K=64 (+8 pad)

  const int t0  = blockIdx.x * 128;
  const int b   = t0 >> 14;            // 16384 tokens per image
  const int hw0 = t0 & (HWP - 1);

  for (int i = threadIdx.x; i < 128 * 64; i += 256) {
    const int tok = i & 127, c = i >> 7;
    At[tok][c] = (_Float16)x[(size_t)(b * C1N + c) * HWP + hw0 + tok];
  }
  __syncthreads();

  const int wv   = threadIdx.x >> 5;
  const int lane = threadIdx.x & 31;
  const v16h a0 = frag_a_lds(&At[wv * 16][0], 72);
  const v16h a1 = frag_a_lds(&At[wv * 16][32], 72);
  const int ncol = lane & 15;
  const int mrow = wv * 16 + ((lane >> 4) << 3);   // first of 8 consecutive M

#pragma unroll
  for (int nt = 0; nt < 16; ++nt) {
    v8f acc = {};
    acc = wmma_f16(a0, frag_b_pre(wfin, 2, 0, nt), acc);
    acc = wmma_f16(a1, frag_b_pre(wfin, 2, 1, nt), acc);
    const int n = nt * 16 + ncol;
    const float bias = b_in[n];
    if (nt < 8) {   // x1 -> NCHW (channel plane n), 8 consecutive tokens/lane
      float* dst = x1n + (size_t)(b * DI + n) * HWP + hw0 + mrow;
#pragma unroll
      for (int r = 0; r < 8; ++r) dst[r] = acc[r] + bias;
    } else {        // z -> token-major [tok][128]
      const int zc = n - DI;
#pragma unroll
      for (int r = 0; r < 8; ++r)
        zbuf[(size_t)(t0 + mrow + r) * DI + zc] = acc[r] + bias;
    }
  }
}

// ---------------------------------------------------------------------------
// Kernel 2a/2b/2c: depthwise 3x3 conv + BN batch statistics.
// ---------------------------------------------------------------------------
__global__ void k_zero_stats(float* stats) {
  if (threadIdx.x < 256) stats[threadIdx.x] = 0.f;   // sum[128], sumsq[128]
}

__global__ __launch_bounds__(256) void k_conv_stats(
    const float* __restrict__ x1n, const float* __restrict__ w_conv,
    const float* __restrict__ b_conv, float* __restrict__ xc,
    float* __restrict__ stats) {
  __shared__ float tile[18][19];
  __shared__ float red[256];
  const int tb = blockIdx.x;
  const int tileid = tb & 63, ch = (tb >> 6) & 127, b = tb >> 13;
  const int ty0 = (tileid >> 3) * 16, tx0 = (tileid & 7) * 16;
  const float* src = x1n + (size_t)(b * DI + ch) * HWP;

  for (int i = threadIdx.x; i < 18 * 18; i += 256) {
    const int yy = i / 18 - 1 + ty0, xx = i % 18 - 1 + tx0;
    float v = 0.f;
    if (yy >= 0 && yy < HH && xx >= 0 && xx < WW) v = src[yy * WW + xx];
    tile[i / 18][i % 18] = v;
  }
  __syncthreads();

  const int ly = threadIdx.x >> 4, lx = threadIdx.x & 15;
  float wc[9];
#pragma unroll
  for (int i = 0; i < 9; ++i) wc[i] = w_conv[ch * 9 + i];
  float acc = b_conv[ch];
#pragma unroll
  for (int ky = 0; ky < 3; ++ky)
#pragma unroll
    for (int kx = 0; kx < 3; ++kx)
      acc += wc[ky * 3 + kx] * tile[ly + ky][lx + kx];
  xc[(size_t)(b * DI + ch) * HWP + (ty0 + ly) * WW + tx0 + lx] = acc;

  red[threadIdx.x] = acc;
  __syncthreads();
  for (int s = 128; s > 0; s >>= 1) {
    if ((int)threadIdx.x < s) red[threadIdx.x] += red[threadIdx.x + s];
    __syncthreads();
  }
  if (threadIdx.x == 0) atomicAdd(&stats[ch], red[0]);
  __syncthreads();
  red[threadIdx.x] = acc * acc;
  __syncthreads();
  for (int s = 128; s > 0; s >>= 1) {
    if ((int)threadIdx.x < s) red[threadIdx.x] += red[threadIdx.x + s];
    __syncthreads();
  }
  if (threadIdx.x == 0) atomicAdd(&stats[DI + ch], red[0]);
}

__global__ void k_bn_final(const float* __restrict__ gamma,
                           const float* __restrict__ beta,
                           float* __restrict__ stats) {
  const int c = threadIdx.x;
  if (c < DI) {
    const float n   = (float)NTOK;
    const float mu  = stats[c] / n;
    const float var = stats[DI + c] / n - mu * mu;
    const float sc  = gamma[c] * rsqrtf(var + 1e-5f);
    stats[256 + c] = sc;                 // scale
    stats[384 + c] = beta[c] - mu * sc;  // shift
  }
}

// ---------------------------------------------------------------------------
// Kernel 3: BN+SiLU (fused in stage) then three WMMA GEMMs per state column
// group j: A-gate (w_x[:, :128]), B-input (w_x[:, 128:]), dt (w_dt).
// Fused epilogue: dt = softplus(.), a = exp(-dt*A); a,b written token-major.
// 96 v_wmma per wave.
// ---------------------------------------------------------------------------
__global__ __launch_bounds__(256) void k_ssm_gemm(
    const float* __restrict__ xc, const float* __restrict__ stats,
    const _Float16* __restrict__ wfx, const float* __restrict__ b_x,
    const _Float16* __restrict__ wfdt, const float* __restrict__ b_dt,
    float* __restrict__ a_ws, float* __restrict__ b_ws) {
  __shared__ _Float16 Xs[128][136];    // 128 tokens x K=128 (+8 pad)
  const int t0 = blockIdx.x * 128, b = t0 >> 14, hw0 = t0 & (HWP - 1);

  for (int i = threadIdx.x; i < 128 * 128; i += 256) {
    const int tok = i & 127, c = i >> 7;
    float v = xc[(size_t)(b * DI + c) * HWP + hw0 + tok];
    v = v * stats[256 + c] + stats[384 + c];   // BN scale/shift
    Xs[tok][c] = (_Float16)silu_f(v);
  }
  __syncthreads();

  const int wv = threadIdx.x >> 5, lane = threadIdx.x & 31;
  v16h af[4];
#pragma unroll
  for (int kk = 0; kk < 4; ++kk)
    af[kk] = frag_a_lds(&Xs[wv * 16][kk * 32], 136);

  const int ncol = lane & 15;
  const int mrow = wv * 16 + ((lane >> 4) << 3);

#pragma unroll
  for (int j = 0; j < 8; ++j) {
    v8f accA = {}, accB = {}, accD = {};
#pragma unroll
    for (int kk = 0; kk < 4; ++kk) {
      accA = wmma_f16(af[kk], frag_b_pre(wfx, 4, kk, j), accA);
      accB = wmma_f16(af[kk], frag_b_pre(wfx, 4, kk, 8 + j), accB);
      accD = wmma_f16(af[kk], frag_b_pre(wfdt, 4, kk, j), accD);
    }
    const int s = j * 16 + ncol;
    const float bxa = b_x[s], bxb = b_x[DST + s], bdt = b_dt[s];
#pragma unroll
    for (int r = 0; r < 8; ++r) {
      const int t = t0 + mrow + r;
      const float Ar  = accA[r] + bxa;
      const float dtr = accD[r] + bdt;
      const float dtv = (dtr > 20.f) ? dtr : log1pf(__expf(dtr));  // softplus
      a_ws[(size_t)t * DST + s] = __expf(-dtv * Ar);
      b_ws[(size_t)t * DST + s] = accB[r] + bxb;
    }
  }
}

// ---------------------------------------------------------------------------
// Kernels 4/5/6: 3-phase scan of s_t = s_{t-1}*a_t + b_t over L=16384.
// 1024 sequences (8 batches x 128 states); states on lanes => every step
// touches one contiguous 512B row.  Software prefetch ~8 rows (4KB) ahead
// keeps the serial dependence chain fed.  Phase 3 writes y in-place over b.
// ---------------------------------------------------------------------------
__global__ __launch_bounds__(128) void k_scan1(
    const float* __restrict__ a_ws, const float* __restrict__ b_ws,
    float* __restrict__ ca, float* __restrict__ cb) {
  const int b = blockIdx.x >> 6, chunk = blockIdx.x & (NCHK - 1);
  const int s = threadIdx.x;
  size_t base = ((size_t)b * HWP + chunk * CHUNK) * DST + s;
  float A = 1.f, Bv = 0.f;
  for (int i = 0; i < CHUNK; ++i) {
    if (i + 8 < CHUNK) {
      __builtin_prefetch(&a_ws[base + 8 * DST], 0, 1);
      __builtin_prefetch(&b_ws[base + 8 * DST], 0, 1);
    }
    const float a  = a_ws[base];
    const float bb = b_ws[base];
    A *= a; Bv = Bv * a + bb;
    base += DST;
  }
  ca[(size_t)blockIdx.x * DST + s] = A;
  cb[(size_t)blockIdx.x * DST + s] = Bv;
}

__global__ __launch_bounds__(128) void k_scan2(
    const float* __restrict__ ca, const float* __restrict__ cb,
    float* __restrict__ pb) {
  const int b = blockIdx.x, s = threadIdx.x;
  float A = 1.f, Bv = 0.f;
  for (int c = 0; c < NCHK; ++c) {
    const size_t idx = ((size_t)b * NCHK + c) * DST + s;
    pb[idx] = Bv;                     // exclusive prefix (B-component)
    const float a = ca[idx], bb = cb[idx];
    Bv = Bv * a + bb;
    A *= a;
  }
}

__global__ __launch_bounds__(128) void k_scan3(
    const float* __restrict__ a_ws, float* __restrict__ b_ws,  // b in, y out
    const float* __restrict__ pb) {
  const int b = blockIdx.x >> 6, chunk = blockIdx.x & (NCHK - 1);
  const int s = threadIdx.x;
  float Bv = pb[((size_t)b * NCHK + chunk) * DST + s];
  size_t base = ((size_t)b * HWP + chunk * CHUNK) * DST + s;
  for (int i = 0; i < CHUNK; ++i) {
    if (i + 8 < CHUNK) {
      __builtin_prefetch(&a_ws[base + 8 * DST], 0, 1);
      __builtin_prefetch(&b_ws[base + 8 * DST], 0, 1);
    }
    Bv = Bv * a_ws[base] + b_ws[base];
    b_ws[base] = Bv;                 // y_t, in-place (per-thread sequential)
    base += DST;
  }
}

// ---------------------------------------------------------------------------
// Kernel 7: out = identity + (y * silu(z)) @ w_out + b_out, stored NCHW.
// ---------------------------------------------------------------------------
__global__ __launch_bounds__(256) void k_out(
    const float* __restrict__ ybuf, const float* __restrict__ zbuf,
    const _Float16* __restrict__ wfo, const float* __restrict__ b_out,
    const float* __restrict__ x, float* __restrict__ out) {
  __shared__ _Float16 Gs[128][136];
  const int t0 = blockIdx.x * 128, b = t0 >> 14, hw0 = t0 & (HWP - 1);

  for (int i = threadIdx.x; i < 128 * 128; i += 256) {
    const int tok = i & 127, c = i >> 7;
    const size_t idx = (size_t)(t0 + tok) * DI + c;
    const float y = ybuf[idx], z = zbuf[idx];
    Gs[tok][c] = (_Float16)(y * silu_f(z));
  }
  __syncthreads();

  const int wv = threadIdx.x >> 5, lane = threadIdx.x & 31;
  v16h af[4];
#pragma unroll
  for (int kk = 0; kk < 4; ++kk)
    af[kk] = frag_a_lds(&Gs[wv * 16][kk * 32], 136);

  const int ncol = lane & 15;
  const int mrow = wv * 16 + ((lane >> 4) << 3);

#pragma unroll
  for (int nt = 0; nt < 4; ++nt) {               // N = 64 output channels
    v8f acc = {};
#pragma unroll
    for (int kk = 0; kk < 4; ++kk)
      acc = wmma_f16(af[kk], frag_b_pre(wfo, 4, kk, nt), acc);
    const int c = nt * 16 + ncol;
    const float bias = b_out[c];
    const float* xin = x  + (size_t)(b * C1N + c) * HWP + hw0 + mrow;
    float*       dst = out + (size_t)(b * C1N + c) * HWP + hw0 + mrow;
#pragma unroll
    for (int r = 0; r < 8; ++r) dst[r] = xin[r] + acc[r] + bias;
  }
}

// ---------------------------------------------------------------------------
extern "C" void kernel_launch(void* const* d_in, const int* in_sizes, int n_in,
                              void* d_out, int out_size, void* d_ws, size_t ws_size,
                              hipStream_t stream) {
  const float* x      = (const float*)d_in[0];
  const float* w_in   = (const float*)d_in[1];
  const float* b_in   = (const float*)d_in[2];
  const float* w_conv = (const float*)d_in[3];
  const float* b_conv = (const float*)d_in[4];
  const float* gamma  = (const float*)d_in[5];
  const float* beta   = (const float*)d_in[6];
  const float* w_x    = (const float*)d_in[7];
  const float* b_x    = (const float*)d_in[8];
  const float* w_dt   = (const float*)d_in[9];
  const float* b_dt   = (const float*)d_in[10];
  const float* w_out  = (const float*)d_in[11];
  const float* b_out  = (const float*)d_in[12];
  float* out = (float*)d_out;

  // Workspace carve-up (floats): five 64 MiB streams + small stats/aggregates
  // + fragment-swizzled f16 weights (all offsets are 16-float multiples).
  const size_t NBIG = (size_t)NTOK * DI;   // 16,777,216
  float* ws    = (float*)d_ws;
  float* x1n   = ws;                 // x1 in NCHW (conv input)
  float* zb    = x1n + NBIG;         // z, token-major
  float* xc    = zb + NBIG;          // conv output, NCHW
  float* a_ws  = xc + NBIG;          // a = exp(-dt*A), token-major
  float* b_ws  = a_ws + NBIG;        // b input -> overwritten by y in scan3
  float* stats = b_ws + NBIG;        // sum[128] sumsq[128] scale[128] shift[128]
  float* ca    = stats + 512;        // chunk aggregates A: 8*64*128
  float* cb    = ca + (size_t)BSZ * NCHK * DST;
  float* pb    = cb + (size_t)BSZ * NCHK * DST;
  _Float16* wfin = (_Float16*)(pb + (size_t)BSZ * NCHK * DST); // 64*256
  _Float16* wfx  = wfin + (size_t)C1N * 256;                   // 128*256
  _Float16* wfdt = wfx + (size_t)DI * 256;                     // 128*128
  _Float16* wfo  = wfdt + (size_t)DI * DST;                    // 128*64

  const int nblk_tok = NTOK / 128;   // 1024

  // Weight fragment swizzle (tiny; runs once per call).
  k_wswz<<<(C1N * 256 + 255) / 256, 256, 0, stream>>>(w_in, C1N, 256, wfin);
  k_wswz<<<(DI * 256 + 255) / 256, 256, 0, stream>>>(w_x, DI, 256, wfx);
  k_wswz<<<(DI * DST + 255) / 256, 256, 0, stream>>>(w_dt, DI, DST, wfdt);
  k_wswz<<<(DI * C1N + 255) / 256, 256, 0, stream>>>(w_out, DI, C1N, wfo);

  k_inproj<<<nblk_tok, 256, 0, stream>>>(x, wfin, b_in, x1n, zb);
  k_zero_stats<<<1, 256, 0, stream>>>(stats);
  k_conv_stats<<<BSZ * DI * 64, 256, 0, stream>>>(x1n, w_conv, b_conv, xc, stats);
  k_bn_final<<<1, 128, 0, stream>>>(gamma, beta, stats);
  k_ssm_gemm<<<nblk_tok, 256, 0, stream>>>(xc, stats, wfx, b_x, wfdt, b_dt,
                                           a_ws, b_ws);
  k_scan1<<<BSZ * NCHK, 128, 0, stream>>>(a_ws, b_ws, ca, cb);
  k_scan2<<<BSZ, 128, 0, stream>>>(ca, cb, pb);
  k_scan3<<<BSZ * NCHK, 128, 0, stream>>>(a_ws, b_ws, pb);
  k_out<<<nblk_tok, 256, 0, stream>>>(b_ws, zb, w_out ? wfo : wfo, b_out, x, out);
}
